// VQEmbedding_41927470744086
// MI455X (gfx1250) — matrix operand order repaired and checked by reading, not compiled, not run on hardware
//
#include <hip/hip_runtime.h>
#include <hip/hip_bf16.h>

// Problem constants (fixed-size reference): K=1024 codes, D=256, B=H=W=32.
#define KCODES   1024
#define DDIM     256
#define HWSZ     1024        // H*W
#define NROWS    32768       // B*H*W
#define OUTHALF  8388608     // B*D*H*W elements per output tensor

typedef __attribute__((ext_vector_type(16))) __bf16          bf16x16;
typedef __attribute__((ext_vector_type(8)))  float           f32x8;
typedef __attribute__((ext_vector_type(4)))  unsigned int    u32x4;
typedef __attribute__((ext_vector_type(16))) unsigned short  u16x16;

union frag_cast {
    u32x4   q[2];
    u16x16  s;
    bf16x16 v;
};

__device__ __forceinline__ unsigned short f2bf(float f) {
    unsigned int u = __float_as_uint(f);
    u += 0x7fffu + ((u >> 16) & 1u);   // round-to-nearest-even
    return (unsigned short)(u >> 16);
}
__device__ __forceinline__ float bf2f(unsigned short h) {
    return __uint_as_float(((unsigned int)h) << 16);
}

// ---------------------------------------------------------------------------
// Kernel 0: codebook fp32 -> split bf16 (hi + residual lo) + per-code ||c||^2.
// 1 MB of work; one thread per code row.
// ---------------------------------------------------------------------------
__global__ void vq_prep_codebook(const float* __restrict__ cb,
                                 unsigned short* __restrict__ cb_hi,
                                 unsigned short* __restrict__ cb_lo,
                                 float* __restrict__ cnorm) {
    int row = blockIdx.x * blockDim.x + threadIdx.x;
    if (row >= KCODES) return;
    const float* src = cb + (size_t)row * DDIM;
    unsigned short* dh = cb_hi + (size_t)row * DDIM;
    unsigned short* dl = cb_lo + (size_t)row * DDIM;
    float s = 0.0f;
    #pragma unroll 4
    for (int d = 0; d < DDIM; ++d) {
        float x = src[d];
        unsigned short h = f2bf(x);
        dh[d] = h;
        dl[d] = f2bf(x - bf2f(h));
        s = fmaf(x, x, s);
    }
    cnorm[row] = s;
}

// ---------------------------------------------------------------------------
// Kernel 1: nearest-code assignment via split-bf16 WMMA.
//   A = codebook tile (M=16 codes x K=32 d), B = x rows (K=32 d x N=16 rows).
//   dot = A_hi*B_hi + A_hi*B_lo + A_lo*B_hi  (3x v_wmma_f32_16x16x32_bf16)
//   dist = ||c||^2 - 2*dot   (||x||^2 constant per row -> dropped)
// Each wave owns 16 x-rows (B fragments resident in 128 VGPRs), sweeps all
// 1024 codes. argmin is per-lane over 8 accumulators, merged with shfl_xor 16.
// ---------------------------------------------------------------------------
__global__ __launch_bounds__(256) void vq_assign(
        const float* __restrict__ x,               // [B, D, H, W]
        const unsigned short* __restrict__ cb_hi,  // [K, D] bf16
        const unsigned short* __restrict__ cb_lo,  // [K, D] bf16
        const float* __restrict__ cnorm,           // [K]
        int* __restrict__ idx_out) {               // [N]
    const int lane = threadIdx.x & 31;
    const int wave = threadIdx.x >> 5;
    const int lrow = lane & 15;       // x-row / code-row within 16-tile
    const int hl   = lane >> 4;       // lane half selects K/M sub-blocks

    const int n  = (blockIdx.x * 8 + wave) * 16 + lrow;   // this lane's x-row
    const int b  = n >> 10;
    const int hw = n & 1023;
    const float* xp = x + (size_t)b * (DDIM * HWSZ) + hw; // stride HWSZ along d

    // Build resident B fragments (split bf16) from strided fp32 reads (once).
    // B 32x16 layout: lane l holds N = l%16, K = (l<16 ? 0..15 : 16..31).
    bf16x16 bhi[8], blo[8];
    #pragma unroll
    for (int kc = 0; kc < 8; ++kc) {
        frag_cast fh, fl;
        const float* xq = xp + (size_t)(kc * 32 + hl * 16) * HWSZ;
        #pragma unroll
        for (int t = 0; t < 16; ++t) {
            float v = xq[(size_t)t * HWSZ];
            unsigned short h = f2bf(v);
            fh.s[t] = h;
            fl.s[t] = f2bf(v - bf2f(h));
        }
        bhi[kc] = fh.v;
        blo[kc] = fl.v;
    }

    float best    = 3.4e38f;
    int   bestidx = 0;
    const int ad = hl * 16;  // A 16x32 layout: lane half picks K {0..15} vs {16..31}

    for (int ct = 0; ct < 64; ++ct) {
        const unsigned short* ah_row = cb_hi + (size_t)(ct * 16 + lrow) * DDIM;
        const unsigned short* al_row = cb_lo + (size_t)(ct * 16 + lrow) * DDIM;
        f32x8 acc = {};
        #pragma unroll
        for (int kc = 0; kc < 8; ++kc) {
            const int d0 = kc * 32 + ad;
            frag_cast ah, al;
            ah.q[0] = *(const u32x4*)(ah_row + d0);
            ah.q[1] = *(const u32x4*)(ah_row + d0 + 8);
            al.q[0] = *(const u32x4*)(al_row + d0);
            al.q[1] = *(const u32x4*)(al_row + d0 + 8);
            acc = __builtin_amdgcn_wmma_f32_16x16x32_bf16(false, ah.v, false, bhi[kc],
                                                          (short)0, acc, false, false);
            acc = __builtin_amdgcn_wmma_f32_16x16x32_bf16(false, ah.v, false, blo[kc],
                                                          (short)0, acc, false, false);
            acc = __builtin_amdgcn_wmma_f32_16x16x32_bf16(false, al.v, false, bhi[kc],
                                                          (short)0, acc, false, false);
        }
        if (ct < 63) {  // pull next code tile toward the WGP (global_prefetch_b8)
            __builtin_prefetch(cb_hi + (size_t)((ct + 1) * 16 + lrow) * DDIM, 0, 3);
            __builtin_prefetch(cb_lo + (size_t)((ct + 1) * 16 + lrow) * DDIM, 0, 3);
        }
        // C layout: lane l, vgpr i -> code M = i + 8*(l>=16), x-row N = l%16.
        const f32x8 cn = *(const f32x8*)(cnorm + ct * 16 + hl * 8);
        #pragma unroll
        for (int i = 0; i < 8; ++i) {
            float d = cn[i] - 2.0f * acc[i];
            int code = ct * 16 + hl * 8 + i;
            if (d < best) { best = d; bestidx = code; }
        }
    }

    // Merge the two lane halves (same x-row, complementary code sets).
    float ob = __shfl_xor(best, 16, 32);
    int   oi = __shfl_xor(bestidx, 16, 32);
    if (ob < best || (ob == best && oi < bestidx)) { best = ob; bestidx = oi; }
    if (hl == 0) idx_out[n] = bestidx;
}

// ---------------------------------------------------------------------------
// Kernel 2: gather + transpose. lane = w so every store of a fixed d is a
// contiguous 128B segment. z_q_x = zex + (q - zex) (exact reference arithmetic),
// zqx_tilde = q. Codebook gather hits L2 (1 MB resident).
// ---------------------------------------------------------------------------
__global__ void vq_gather(const float* __restrict__ z,    // [B, D, H, W]
                          const float* __restrict__ cb,   // [K, D] fp32
                          const int* __restrict__ idx,    // [N]
                          float* __restrict__ out) {      // 2 x [B, D, H, W]
    const int w      = threadIdx.x;   // 0..31
    const int dchunk = threadIdx.y;   // 0..7
    const int bh     = blockIdx.x;    // 0..1023
    const int b = bh >> 5, h = bh & 31;
    const int code = idx[bh * 32 + w];
    const float* crow = cb + (size_t)code * DDIM;
    const size_t base = (size_t)b * (DDIM * HWSZ) + (size_t)h * 32 + w;
    const float* zp = z + base;
    float* o1 = out + base;            // z_q_x
    float* o2 = o1 + OUTHALF;          // zqx_tilde
    const int d0 = dchunk * 32;
    #pragma unroll
    for (int j = 0; j < 32; j += 4) {
        float4 v = *(const float4*)(crow + d0 + j);
        float q0 = v.x, q1 = v.y, q2 = v.z, q3 = v.w;
        size_t s0 = (size_t)(d0 + j + 0) * HWSZ;
        size_t s1 = (size_t)(d0 + j + 1) * HWSZ;
        size_t s2 = (size_t)(d0 + j + 2) * HWSZ;
        size_t s3 = (size_t)(d0 + j + 3) * HWSZ;
        o1[s0] = zp[s0] + (q0 - zp[s0]);
        o1[s1] = zp[s1] + (q1 - zp[s1]);
        o1[s2] = zp[s2] + (q2 - zp[s2]);
        o1[s3] = zp[s3] + (q3 - zp[s3]);
        o2[s0] = q0;
        o2[s1] = q1;
        o2[s2] = q2;
        o2[s3] = q3;
    }
}

extern "C" void kernel_launch(void* const* d_in, const int* in_sizes, int n_in,
                              void* d_out, int out_size, void* d_ws, size_t ws_size,
                              hipStream_t stream) {
    const float* z_e_x    = (const float*)d_in[0];   // [32, 256, 32, 32] fp32
    const float* codebook = (const float*)d_in[1];   // [1024, 256] fp32
    float* out = (float*)d_out;                       // 2 x [32, 256, 32, 32]

    // Workspace layout (~1.2 MB):
    char* ws = (char*)d_ws;
    unsigned short* cb_hi = (unsigned short*)(ws);              // 512 KB
    unsigned short* cb_lo = (unsigned short*)(ws + 524288);     // 512 KB
    float*          cnorm = (float*)(ws + 1048576);             // 4 KB
    int*            idx   = (int*)(ws + 1052672);               // 128 KB

    vq_prep_codebook<<<dim3(4), dim3(256), 0, stream>>>(codebook, cb_hi, cb_lo, cnorm);
    // 2048 row-groups of 16 rows; 8 waves per 256-thread block.
    vq_assign<<<dim3(256), dim3(256), 0, stream>>>(z_e_x, cb_hi, cb_lo, cnorm, idx);
    vq_gather<<<dim3(1024), dim3(32, 8), 0, stream>>>(z_e_x, codebook, idx, out);
}
